// ChannelAttention_56573309224430
// MI455X (gfx1250) — matrix operand order repaired and checked by reading, not compiled, not run on hardware
//
#include <hip/hip_runtime.h>
#include <hip/hip_bf16.h>

// ---------------------------------------------------------------------------
// Shapes fixed by setup_inputs(): B=8, L=S=1024, H=8, E=64. n_tokens read from
// device memory (d_in[6]). Softmax runs in the base-2 domain: Q and bias are
// pre-scaled by SCALE*LOG2E so the inner loop needs no per-element multiplies.
// Row sums are produced by an all-ones WMMA; row max needs only one xor-16
// shuffle per chunk (scores are re-read from LDS in row-per-lane A-layout).
// ---------------------------------------------------------------------------
constexpr int B = 8, L = 1024, H = 8, E = 64, S = 1024;
constexpr float SCALE  = 0.125f;                      // 1/sqrt(64)
constexpr float LOG2E  = 1.4426950408889634f;
constexpr float QSCALE = SCALE * LOG2E;               // folded into Q + bias
constexpr float MASKED = -1e30f;                      // exp2 -> 0

constexpr int ROWSTRIDE = 40;                         // f32 LDS row pitch (16B-aligned, bank-spread)
constexpr int LDS_PER_WAVE = 16 * ROWSTRIDE + 16;     // score tile + alpha strip

typedef _Float16 v16h __attribute__((ext_vector_type(16)));
typedef _Float16 v8h  __attribute__((ext_vector_type(8)));
typedef _Float16 v2h  __attribute__((ext_vector_type(2)));
typedef float    v8f  __attribute__((ext_vector_type(8)));
typedef float    v4f  __attribute__((ext_vector_type(4)));

// ---------------------------------------------------------------------------
// Prep 1: [B,L,H,E] fp32 -> [B,H,L,E] f16 with interleaved RoPE on dims [0,32)
// theta_j = 10000^(-j/16), angle = pos * theta_j, pairs (2j, 2j+1).
// `mul` = SCALE*LOG2E for Q, 1.0 for K (scale folded into the dot product).
// ---------------------------------------------------------------------------
__global__ void rope_to_f16(const float* __restrict__ src, _Float16* __restrict__ dst,
                            float mul) {
  int idx = blockIdx.x * blockDim.x + threadIdx.x;            // over B*L*H*(E/2)
  int j = idx & (E / 2 - 1);
  int t = idx >> 5;                                           // /(E/2)=32
  int h = t % H;  t /= H;
  int l = t % L;
  int b = t / L;
  int e = 2 * j;
  const float* p = src + ((((size_t)b * L + l) * H + h) * E + e);
  float x0 = p[0], x1 = p[1];
  if (j < E / 4) {                                            // RoPE'd half (e < 32)
    float theta = __powf(10000.0f, -(float)j * (1.0f / (float)(E / 4)));
    float ang = (float)l * theta;
    float sn, cs;
    __sincosf(ang, &sn, &cs);
    float r0 = x0 * cs - x1 * sn;
    float r1 = x1 * cs + x0 * sn;
    x0 = r0; x1 = r1;
  }
  v2h pk; pk[0] = (_Float16)(x0 * mul); pk[1] = (_Float16)(x1 * mul);
  *(v2h*)(dst + ((((size_t)b * H + h) * L + l) * E + e)) = pk;
}

// ---------------------------------------------------------------------------
// Prep 2: values [B,S,H,E] fp32 -> Vt [B,H,E,S] f16 (coalesced writes along S).
// ---------------------------------------------------------------------------
__global__ void v_to_f16_t(const float* __restrict__ v, _Float16* __restrict__ vt) {
  int idx = blockIdx.x * blockDim.x + threadIdx.x;            // over B*H*E*S
  int s = idx & (S - 1);
  int t = idx >> 10;                                          // /S
  int e = t % E;  t /= E;
  int h = t % H;
  int b = t / H;
  float val = v[((((size_t)b * S + s) * H + h) * E) + e];
  vt[idx] = (_Float16)val;
}

// ---------------------------------------------------------------------------
// Fused flash attention. One wave32 per 16-row Q tile; S swept in 32-key
// chunks. 9 v_wmma per chunk (4 QK^T + 1 row-sum + 4 PV).
// ---------------------------------------------------------------------------
__launch_bounds__(256)
__global__ void attn_kernel(const _Float16* __restrict__ Qh,
                            const _Float16* __restrict__ Kh,
                            const _Float16* __restrict__ Vt,
                            const float*    __restrict__ bias_emb,
                            const int*      __restrict__ nt_ptr,
                            float*          __restrict__ out) {
  const int nt = *nt_ptr;                                     // n_tokens (64)
  __shared__ float ldsBuf[8][LDS_PER_WAVE];                   // 2.6 KB per wave

  const int wave = threadIdx.x >> 5;
  const int lane = threadIdx.x & 31;
  const int ln   = lane & 15;                                 // column / row-in-tile
  const int hi   = lane >> 4;                                 // lane half

  int tileId = blockIdx.x * 8 + wave;                         // over B*H*(L/16)
  const int qt = tileId % (L / 16);
  int t = tileId / (L / 16);
  const int h = t % H;
  const int b = t / H;
  const int m0 = qt * 16;

  const _Float16* Qb = Qh + (size_t)(b * H + h) * L * E;
  const _Float16* Kb = Kh + (size_t)(b * H + h) * S * E;
  const _Float16* Vb = Vt + (size_t)(b * H + h) * E * S;
  const float bias0 = bias_emb[h]     * QSCALE;               // base-2 domain
  const float bias1 = bias_emb[H + h] * QSCALE;
  float* Ptile = ldsBuf[wave];                                // 16 x ROWSTRIDE f32
  float* Astrip = Ptile + 16 * ROWSTRIDE;                     // 16 f32 alpha strip

  // ---- Q tile as two A-operands (16x32 f16 layout: lane=row, K split by half)
  const int arow = m0 + ln;
  const int aoff = hi * 8;                                    // K sub-offset per half
  v16h qa[2];
#pragma unroll
  for (int ec = 0; ec < 2; ++ec) {
    v8h lo = *(const v8h*)(Qb + (size_t)arow * E + ec * 32 + aoff);
    v8h hi8 = *(const v8h*)(Qb + (size_t)arow * E + ec * 32 + 16 + aoff);
#pragma unroll
    for (int i = 0; i < 8; ++i) { qa[ec][i] = lo[i]; qa[ec][i + 8] = hi8[i]; }
  }
  v16h onesB;                                                 // all-ones B => row sums
#pragma unroll
  for (int i = 0; i < 16; ++i) onesB[i] = (_Float16)1.0f;

  // ---- per-lane query-row state (A-side: lane ln owns row m0+ln)
  const int tq = (m0 + ln) % nt;                              // init-only div
  const int vq = (m0 + ln) / nt;
  float mold = MASKED;
  v8f acc[4] = {};                                            // 16x64 fp32 accum (C layout)
  v8f lacc = {};                                              // row-sum accum (C layout)

  const int tqmax = ((m0 % nt) + 15 < nt) ? (m0 % nt) + 15 : nt - 1;

  // incremental key-side counters: k-slot groups g=0 -> k=aoff.., g=1 -> k=16+aoff..
  int tkb[2], vkb[2];
#pragma unroll
  for (int g = 0; g < 2; ++g) {
    int s = g * 16 + aoff;
    tkb[g] = s % nt; vkb[g] = s / nt;                         // init-only div
  }
  int rk = 0;                                                 // s0 % nt

  for (int s0 = 0; s0 < S; s0 += 32) {
    // wave-uniform skip of fully masked chunks (min tk > max tq)
    int mintk = (rk + 31 < nt) ? rk : 0;
    if (mintk <= tqmax) {
      // ---- scores: 16x32 tile = 2 N-halves, K=64 split into 2 chunks of 32
      v8f st[2] = {};
#pragma unroll
      for (int nh = 0; nh < 2; ++nh) {
        int srow = s0 + nh * 16 + ln;                         // B column = key row
#pragma unroll
        for (int ec = 0; ec < 2; ++ec) {
          v8h lo = *(const v8h*)(Kb + (size_t)srow * E + ec * 32 + hi * 8);
          v8h hi8 = *(const v8h*)(Kb + (size_t)srow * E + ec * 32 + 16 + hi * 8);
          v16h kb;
#pragma unroll
          for (int i = 0; i < 8; ++i) { kb[i] = lo[i]; kb[i + 8] = hi8[i]; }
          st[nh] = __builtin_amdgcn_wmma_f32_16x16x32_f16(
              false, qa[ec], false, kb, (short)0, st[nh], false, false);
        }
      }

      // ---- spill raw scores (C layout) to LDS
#pragma unroll
      for (int nh = 0; nh < 2; ++nh)
#pragma unroll
        for (int r = 0; r < 8; ++r)
          Ptile[(r + 8 * hi) * ROWSTRIDE + nh * 16 + ln] = st[nh][r];
      asm volatile("s_wait_dscnt 0x0" ::: "memory");

      // ---- re-read in A-layout: lane ln owns row ln, k-slots aoff..+7, 16+aoff..+7
      float sr[16];
      {
        const float* rowp = Ptile + ln * ROWSTRIDE + aoff;
        v4f a0 = *(const v4f*)(rowp);
        v4f a1 = *(const v4f*)(rowp + 4);
        v4f a2 = *(const v4f*)(rowp + 16);
        v4f a3 = *(const v4f*)(rowp + 20);
#pragma unroll
        for (int i = 0; i < 4; ++i) {
          sr[i] = a0[i]; sr[i + 4] = a1[i]; sr[i + 8] = a2[i]; sr[i + 12] = a3[i];
        }
      }

      // ---- bias + mask, local to the lane (tq/vq are scalars here)
#pragma unroll
      for (int g = 0; g < 2; ++g)
#pragma unroll
        for (int o = 0; o < 8; ++o) {
          int i = g * 8 + o;
          int tk0 = tkb[g] + o;                               // wraps at most once (nt>=8)
          bool c = tk0 >= nt;
          int tke = c ? tk0 - nt : tk0;
          int vke = vkb[g] + (c ? 1 : 0);
          float val = sr[i] + ((vke == vq) ? bias1 : bias0);
          sr[i] = (tke > tq) ? MASKED : val;
        }

      // ---- row max: 15 local max + one xor-16 merge with partner half-row
      float mx = sr[0];
#pragma unroll
      for (int i = 1; i < 16; ++i) mx = fmaxf(mx, sr[i]);
      mx = fmaxf(mx, __shfl_xor(mx, 16, 32));
      float mnew = fmaxf(mold, mx);
      float aown = exp2f(mold - mnew);
      mold = mnew;

      // ---- p = exp2(s - m), packed straight into the A operand
      v16h pa;
#pragma unroll
      for (int i = 0; i < 16; ++i) pa[i] = (_Float16)exp2f(sr[i] - mnew);

      // ---- broadcast alpha (lane=row) -> C-layout rows r+8*hi
      Astrip[ln] = aown;                                      // both halves: same value
      asm volatile("s_wait_dscnt 0x0" ::: "memory");
      float alpha[8];
      {
        v4f a0 = *(const v4f*)(Astrip + 8 * hi);
        v4f a1 = *(const v4f*)(Astrip + 8 * hi + 4);
#pragma unroll
        for (int i = 0; i < 4; ++i) { alpha[i] = a0[i]; alpha[i + 4] = a1[i]; }
      }

      // ---- row sums via all-ones WMMA (replaces 32 ds_bpermute butterflies)
      {
        v8f la = lacc;
#pragma unroll
        for (int r = 0; r < 8; ++r) la[r] *= alpha[r];
        lacc = __builtin_amdgcn_wmma_f32_16x16x32_f16(
            false, pa, false, onesB, (short)0, la, false, false);
      }

      // ---- acc = acc*alpha + P(16x32) @ V(32x16) per 16-wide E chunk
#pragma unroll
      for (int ec = 0; ec < 4; ++ec) {
        v8f a = acc[ec];
#pragma unroll
        for (int r = 0; r < 8; ++r) a[r] *= alpha[r];
        const _Float16* vp = Vb + (size_t)(ec * 16 + ln) * S + s0 + hi * 8;
        v8h lo = *(const v8h*)(vp);
        v8h hi8 = *(const v8h*)(vp + 16);
        v16h vb;
#pragma unroll
        for (int i = 0; i < 8; ++i) { vb[i] = lo[i]; vb[i + 8] = hi8[i]; }
        acc[ec] = __builtin_amdgcn_wmma_f32_16x16x32_f16(
            false, pa, false, vb, (short)0, a, false, false);
      }
    }

    // ---- advance counters (runs on skipped chunks too; no div/mod)
    rk += 32;
    while (rk >= nt) rk -= nt;
#pragma unroll
    for (int g = 0; g < 2; ++g) {
      tkb[g] += 32;
      while (tkb[g] >= nt) { tkb[g] -= nt; ++vkb[g]; }
    }
  }

  // ---- normalize (lacc holds row sums replicated across lanes) and write out
  float inv[8];
#pragma unroll
  for (int r = 0; r < 8; ++r) inv[r] = 1.0f / lacc[r];
#pragma unroll
  for (int ec = 0; ec < 4; ++ec)
#pragma unroll
    for (int r = 0; r < 8; ++r) {
      int m = m0 + r + 8 * hi;
      int e = ec * 16 + ln;
      out[(((size_t)b * L + m) * H + h) * E + e] = acc[ec][r] * inv[r];
    }
}

// ---------------------------------------------------------------------------
extern "C" void kernel_launch(void* const* d_in, const int* in_sizes, int n_in,
                              void* d_out, int out_size, void* d_ws, size_t ws_size,
                              hipStream_t stream) {
  (void)in_sizes; (void)n_in; (void)out_size; (void)ws_size;
  const float* q    = (const float*)d_in[0];
  const float* k    = (const float*)d_in[1];
  const float* v    = (const float*)d_in[2];
  const float* bias = (const float*)d_in[3];
  const int* n_tok  = (const int*)d_in[6];
  float* out = (float*)d_out;

  _Float16* Qh = (_Float16*)d_ws;                     // [B,H,L,E] f16 (8 MiB)
  _Float16* Kh = Qh + (size_t)B * H * L * E;          // [B,H,S,E] f16 (8 MiB)
  _Float16* Vt = Kh + (size_t)B * H * S * E;          // [B,H,E,S] f16 (8 MiB)

  {
    int n = B * L * H * (E / 2);
    rope_to_f16<<<n / 256, 256, 0, stream>>>(q, Qh, QSCALE);  // scale folded in
    rope_to_f16<<<n / 256, 256, 0, stream>>>(k, Kh, 1.0f);
  }
  {
    int n = B * H * E * S;
    v_to_f16_t<<<n / 256, 256, 0, stream>>>(v, Vt);
  }
  {
    int tiles = B * H * (L / 16);                     // 4096 wave-tiles
    attn_kernel<<<tiles / 8, 256, 0, stream>>>(Qh, Kh, Vt, bias, n_tok, out);
  }
}
